// NaiveAttention_72000831750648
// MI455X (gfx1250) — matrix operand order repaired and checked by reading, not compiled, not run on hardware
//
#include <hip/hip_runtime.h>
#include <hip/hip_bf16.h>

typedef __attribute__((ext_vector_type(16))) _Float16 v16h;
typedef __attribute__((ext_vector_type(8)))  _Float16 v8h;
typedef __attribute__((ext_vector_type(2)))  _Float16 v2h;
typedef __attribute__((ext_vector_type(8)))  float    v8f;

// Problem constants (from reference setup_inputs)
static constexpr int B = 2;
static constexpr int H = 16;
static constexpr int S = 2048;
static constexpr int D = 64;

static constexpr int QTILE = 16;            // query rows per wave (WMMA M)
static constexpr int KBLK  = 64;            // keys per block iteration
static constexpr int WAVES = 8;             // waves per workgroup
static constexpr int QBLK  = QTILE * WAVES; // 128 query rows per workgroup
static constexpr int KSTR  = 72;            // LDS K row stride (halves), 16B aligned + pad
static constexpr int VSTR  = 72;            // LDS V^T row stride (halves)

// scores are computed in the exp2 domain: fold log2(e)/sqrt(D) into Q
static constexpr float QSCALE = 0.125f * 1.44269504088896340736f;

__global__ __launch_bounds__(256)
void fa_fwd_kernel(const float* __restrict__ qg,
                   const float* __restrict__ kg,
                   const float* __restrict__ vg,
                   const unsigned char* __restrict__ maskg,
                   float* __restrict__ og) {
    __shared__ __align__(16) _Float16 Kl[2][KBLK * KSTR];    // K block, [key][d], f16
    __shared__ __align__(16) _Float16 Vl[2][D * VSTR];       // V block transposed, [d][key]
    __shared__ __align__(16) _Float16 Pl[WAVES * 16 * KBLK]; // per-wave P staging

    const int tid  = threadIdx.x;
    const int wave = tid >> 5;
    const int lane = tid & 31;
    const int col  = lane & 15;              // C/B-layout column
    const int half8 = (lane & 16) ? 8 : 0;   // C-layout: lanes 16-31 hold rows M+8
    const int arow  = lane & 15;             // A-layout row
    const int kofs  = (lane & 16) ? 16 : 0;  // A/B k order chosen here: lanes 16-31 hold K+16

    const int bh = blockIdx.y;
    const int b  = bh / H;
    const int q0 = blockIdx.x * QBLK + wave * QTILE;
    const size_t base  = (size_t)bh * S * D;
    const size_t mbase = (size_t)b * S;

    // ---- Q tile as pre-scaled f16 A-operands (resident for whole kernel) ----
    v16h aq0, aq1;
    {
        const float* qp = qg + base + (size_t)(q0 + arow) * D;
        #pragma unroll
        for (int j = 0; j < 16; ++j) {
            int kk = j + kofs;
            aq0[j] = (_Float16)(qp[kk]      * QSCALE);
            aq1[j] = (_Float16)(qp[32 + kk] * QSCALE);
        }
    }

    v16h vones;
    #pragma unroll
    for (int j = 0; j < 16; ++j) vones[j] = (_Float16)1.0f;

    // ---- running state: wave-shared max, WMMA row-sum + output accumulators ----
    float mrun = -__builtin_inff();
    v8f acc[4], accl;
    v8f vzero = {};
    #pragma unroll
    for (int t = 0; t < 4; ++t) acc[t] = vzero;
    accl = vzero;

    _Float16* Pw = &Pl[wave * 16 * KBLK];

    // Loader mappings (256 threads, 64x64 f32 per matrix per block)
    const int rK = tid >> 2;          // K: key row 0..63
    const int cK = (tid & 3) * 16;    // K: dim start {0,16,32,48}
    const int rV = (tid >> 3) * 2;    // V: key pair 0..62
    const int cV = (tid & 7) * 8;     // V: dim start {0,..,56}

    const int num_kb = (blockIdx.x + 1) * (QBLK / KBLK);

    // staged registers for double buffering
    float4 kr0, kr1, kr2, kr3, va0, va1, vb0, vb1;
    auto gload = [&](int kb) {
        const float* kp = kg + base + (size_t)(kb * KBLK + rK) * D + cK;
        kr0 = *(const float4*)(kp);
        kr1 = *(const float4*)(kp + 4);
        kr2 = *(const float4*)(kp + 8);
        kr3 = *(const float4*)(kp + 12);
        const float* vp = vg + base + (size_t)(kb * KBLK + rV) * D + cV;
        va0 = *(const float4*)(vp);
        va1 = *(const float4*)(vp + 4);
        vb0 = *(const float4*)(vp + D);
        vb1 = *(const float4*)(vp + D + 4);
    };

    gload(0);

    for (int kb = 0; kb < num_kb; ++kb) {
        const int cur = kb & 1;
        _Float16* Kb = &Kl[cur][0];
        _Float16* Vb = &Vl[cur][0];

        // ---- store staged regs into LDS (f16; V transposed, key pairs packed) ----
        {
            v8h h0, h1;
            h0[0]=(_Float16)kr0.x; h0[1]=(_Float16)kr0.y; h0[2]=(_Float16)kr0.z; h0[3]=(_Float16)kr0.w;
            h0[4]=(_Float16)kr1.x; h0[5]=(_Float16)kr1.y; h0[6]=(_Float16)kr1.z; h0[7]=(_Float16)kr1.w;
            h1[0]=(_Float16)kr2.x; h1[1]=(_Float16)kr2.y; h1[2]=(_Float16)kr2.z; h1[3]=(_Float16)kr2.w;
            h1[4]=(_Float16)kr3.x; h1[5]=(_Float16)kr3.y; h1[6]=(_Float16)kr3.z; h1[7]=(_Float16)kr3.w;
            *(v8h*)&Kb[rK * KSTR + cK]     = h0;
            *(v8h*)&Kb[rK * KSTR + cK + 8] = h1;
            float a0[8] = {va0.x, va0.y, va0.z, va0.w, va1.x, va1.y, va1.z, va1.w};
            float a1[8] = {vb0.x, vb0.y, vb0.z, vb0.w, vb1.x, vb1.y, vb1.z, vb1.w};
            #pragma unroll
            for (int i = 0; i < 8; ++i) {
                v2h pr;
                pr[0] = (_Float16)a0[i];
                pr[1] = (_Float16)a1[i];
                *(v2h*)&Vb[(cV + i) * VSTR + rV] = pr;
            }
        }
        __syncthreads();

        // ---- issue next block's global loads (overlap with compute) ----
        if (kb + 1 < num_kb) gload(kb + 1);
        if (kb + 2 < num_kb) {   // global_prefetch of the block after that
            __builtin_prefetch(kg + base + (size_t)((kb + 2) * KBLK + rK) * D + cK, 0, 3);
            __builtin_prefetch(vg + base + (size_t)((kb + 2) * KBLK + rV) * D + cV, 0, 3);
        }

        const bool active = (kb * KBLK) <= (q0 + QTILE - 1);
        if (active) {
            // ---- scores: 4 tiles of 16 keys, K-dim 64 in two chunks ----
            v8f s[4];
            #pragma unroll
            for (int t = 0; t < 4; ++t) {
                const _Float16* krow = &Kb[(t * 16 + col) * KSTR];
                v8f cs = vzero;
                v16h b0 = *(const v16h*)&krow[kofs];
                cs = __builtin_amdgcn_wmma_f32_16x16x32_f16(false, aq0, false, b0,
                                                            (short)0, cs, false, false);
                v16h b1 = *(const v16h*)&krow[32 + kofs];
                cs = __builtin_amdgcn_wmma_f32_16x16x32_f16(false, aq1, false, b1,
                                                            (short)0, cs, false, false);
                s[t] = cs;
            }

            // ---- masking: only on the diagonal block / padded keys (uniform branch) ----
            const int key0 = kb * KBLK + col;
            bool kp[4];
            #pragma unroll
            for (int t = 0; t < 4; ++t) kp[t] = maskg[mbase + key0 + t * 16] != 0;
            const bool lanekeep   = kp[0] && kp[1] && kp[2] && kp[3];
            const bool allkeep    = __all(lanekeep);
            const bool fullcausal = (kb * KBLK + KBLK - 1) <= q0;
            if (!(allkeep && fullcausal)) {
                #pragma unroll
                for (int t = 0; t < 4; ++t) {
                    const int key = key0 + t * 16;
                    #pragma unroll
                    for (int i = 0; i < 8; ++i) {
                        const int row = q0 + i + half8;
                        if (key > row || !kp[t]) s[t][i] = -__builtin_inff();
                    }
                }
            }

            // ---- tile-shared max: local 32-value tree + full-wave shuffle reduce ----
            float bm = s[0][0];
            #pragma unroll
            for (int t = 0; t < 4; ++t)
                #pragma unroll
                for (int i = 0; i < 8; ++i)
                    bm = fmaxf(bm, s[t][i]);
            #pragma unroll
            for (int off = 1; off < 32; off <<= 1)
                bm = fmaxf(bm, __shfl_xor(bm, off, 32));

            const float mn = fmaxf(mrun, bm);
            // wave-uniform by construction -> scalar branch via readfirstlane
            const int resc = __builtin_amdgcn_readfirstlane((mn > mrun) ? 1 : 0);
            const float scl = resc ? __builtin_exp2f(mrun - mn) : 1.0f;
            mrun = mn;

            // ---- probabilities (exp2 domain) -> per-wave LDS staging ----
            #pragma unroll
            for (int i = 0; i < 8; ++i) {
                _Float16* pr = &Pw[(i + half8) * KBLK + col];
                pr[0]  = (_Float16)__builtin_exp2f(s[0][i] - mn);
                pr[16] = (_Float16)__builtin_exp2f(s[1][i] - mn);
                pr[32] = (_Float16)__builtin_exp2f(s[2][i] - mn);
                pr[48] = (_Float16)__builtin_exp2f(s[3][i] - mn);
            }

            // ---- rescale accumulators only when the running max grew ----
            if (resc) {
                #pragma unroll
                for (int t = 0; t < 4; ++t)
                    #pragma unroll
                    for (int i = 0; i < 8; ++i)
                        acc[t][i] *= scl;
                #pragma unroll
                for (int i = 0; i < 8; ++i)
                    accl[i] *= scl;
            }

            asm volatile("" ::: "memory");  // order LDS store->load (same-wave DS in-order)

            // ---- rebuild P as two 16x32 A-operands ----
            const _Float16* prow = &Pw[arow * KBLK];
            v16h ap0 = __builtin_shufflevector(*(const v8h*)&prow[kofs],
                                               *(const v8h*)&prow[kofs + 8],
                                               0,1,2,3,4,5,6,7,8,9,10,11,12,13,14,15);
            v16h ap1 = __builtin_shufflevector(*(const v8h*)&prow[32 + kofs],
                                               *(const v8h*)&prow[40 + kofs],
                                               0,1,2,3,4,5,6,7,8,9,10,11,12,13,14,15);

            // ---- row-sum l via WMMA: accl += P x ones ----
            accl = __builtin_amdgcn_wmma_f32_16x16x32_f16(false, ap0, false, vones,
                                                          (short)0, accl, false, false);
            accl = __builtin_amdgcn_wmma_f32_16x16x32_f16(false, ap1, false, vones,
                                                          (short)0, accl, false, false);

            // ---- O += P x V (4 tiles over D, two 32-key halves) ----
            #pragma unroll
            for (int t = 0; t < 4; ++t) {
                v8f a = acc[t];
                const _Float16* vrow = &Vl[cur][(t * 16 + col) * VSTR];
                v16h bv0 = *(const v16h*)&vrow[kofs];
                a = __builtin_amdgcn_wmma_f32_16x16x32_f16(false, ap0, false, bv0,
                                                           (short)0, a, false, false);
                v16h bv1 = *(const v16h*)&vrow[32 + kofs];
                a = __builtin_amdgcn_wmma_f32_16x16x32_f16(false, ap1, false, bv1,
                                                           (short)0, a, false, false);
                acc[t] = a;
            }
        }
    }

    // ---- normalize by l (from WMMA row-sum accumulator) and write out ----
    #pragma unroll
    for (int i = 0; i < 8; ++i) {
        const float inv = 1.0f / accl[i];
        float* op = og + base + (size_t)(q0 + i + half8) * D;
        #pragma unroll
        for (int t = 0; t < 4; ++t)
            op[t * 16 + col] = acc[t][i] * inv;
    }
}

extern "C" void kernel_launch(void* const* d_in, const int* in_sizes, int n_in,
                              void* d_out, int out_size, void* d_ws, size_t ws_size,
                              hipStream_t stream) {
    const float* q = (const float*)d_in[0];
    const float* k = (const float*)d_in[1];
    const float* v = (const float*)d_in[2];
    const unsigned char* mask = (const unsigned char*)d_in[3];
    float* out = (float*)d_out;

    dim3 grid(S / QBLK, B * H);   // 16 x 32
    dim3 block(WAVES * 32);       // 256 threads = 8 waves
    fa_fwd_kernel<<<grid, block, 0, stream>>>(q, k, v, mask, out);
}
